// SelfAttentionBlock_58918361366561
// MI455X (gfx1250) — matrix operand order repaired and checked by reading, not compiled, not run on hardware
//
#include <hip/hip_runtime.h>

// ---------------------------------------------------------------------------
// SelfAttentionBlock on MI455X (gfx1250): GroupNorm -> QKV -> flash-attention
// -> proj + residual. f16 WMMA (v_wmma_f32_16x16x32_f16) with f32 accum.
// B=4, C=128, H=W=64 -> N=4096 tokens, 36.5 GFLOP; all intermediates (~21 MB
// f16) live in the 192 MB L2. Attention K/V tiles are staged block-wide into
// LDS via CDNA5 async global->LDS DMA (ASYNCcnt), double-buffered to overlap
// the next tile's copy with the current tile's WMMA + online softmax.
// ---------------------------------------------------------------------------

typedef __attribute__((ext_vector_type(16))) _Float16 v16h_t;
typedef __attribute__((ext_vector_type(2)))  _Float16 v2h_t;
typedef __attribute__((ext_vector_type(8)))  float    v8f_t;

#define BATCH 4
#define CH    128
#define SEQ   4096                 // H*W
#define ROWS  (BATCH * SEQ)        // 16384 flat token rows
#define CPG   4                    // channels per group (128/32)
#define GN_EPS 1e-5f
#define ATTN_SCALE 0.08838834764831845f   // 1/sqrt(128)

__device__ __forceinline__ v8f_t wmma_f16(v16h_t a, v16h_t b, v8f_t c) {
  return __builtin_amdgcn_wmma_f32_16x16x32_f16(false, a, false, b,
                                                (short)0, c, false, false);
}

// A-fragment (16x32 f16, M x K): lane holds row = row0 + (lane&15);
// lanes 0-15 hold K=k0..k0+15, lanes 16-31 hold K=k0+16..k0+31 (matches B).
__device__ __forceinline__ v16h_t load_a16(const _Float16* base, int ld,
                                           int row0, int k0, int lane) {
  const int r = row0 + (lane & 15);
  const int khalf = (lane & 16) ? 16 : 0;
  const _Float16* p = base + (long)r * ld + k0 + khalf;
  v16h_t a;
#pragma unroll
  for (int v = 0; v < 8; ++v) {
    const int kk = 2 * v;
    v2h_t t = *(const v2h_t*)(p + kk);
    a[2 * v] = t[0];
    a[2 * v + 1] = t[1];
  }
  return a;
}

// B-fragment (32x16 f16, K x N) for A * W^T with W row-major [N][K]:
// lane holds col n = n0 + (lane&15); lanes 0-15 hold K=k0..k0+15,
// lanes 16-31 hold K=k0+16..k0+31, pairs contiguous along K.
__device__ __forceinline__ v16h_t load_bt16(const _Float16* w, int ld,
                                            int n0, int k0, int lane) {
  const int n = n0 + (lane & 15);
  const int kh = (lane & 16) ? 16 : 0;
  const _Float16* p = w + (long)n * ld + k0 + kh;
  v16h_t b;
#pragma unroll
  for (int v = 0; v < 8; ++v) {
    v2h_t t = *(const v2h_t*)(p + 2 * v);
    b[2 * v] = t[0];
    b[2 * v + 1] = t[1];
  }
  return b;
}

// CDNA5 async DMA: 16 bytes global -> LDS per lane, tracked by ASYNCcnt.
// VDST operand carries the wave-relative LDS byte address (low 32 bits of the
// generic shared pointer -- the LDS aperture lives in addr[63:32]).
__device__ __forceinline__ void async_cp16(unsigned lds_addr,
                                           const _Float16* gsrc) {
  asm volatile("global_load_async_to_lds_b128 %0, %1, off"
               :: "v"(lds_addr), "v"(gsrc)
               : "memory");
}
__device__ __forceinline__ void wait_async0() {
  asm volatile("s_wait_asynccnt 0x0" ::: "memory");
}

// Stage one 32-key tile: K tile (32 rows x 128 f16 = 8KB, contiguous) and
// V^T tile (128 rows x 32 f16 = 8KB, strided rows). 4 async b128 per thread.
__device__ __forceinline__ void stage_tile(const _Float16* kb, const _Float16* vb,
                                           const _Float16* ks, const _Float16* vs,
                                           int kt, int tid) {
  const _Float16* gk = kb + (long)kt * 32 * CH;
  const unsigned lk = (unsigned)(size_t)ks;
  const unsigned lv = (unsigned)(size_t)vs;
#pragma unroll
  for (int i = 0; i < 2; ++i) {
    const int e = tid + i * 256;                 // 16-byte chunk id 0..511
    async_cp16(lk + e * 16, gk + e * 8);
  }
#pragma unroll
  for (int i = 0; i < 2; ++i) {
    const int ch = tid + i * 256;                // 0..511
    const int row = ch >> 2;                     // V^T channel row
    const int off = (ch & 3) * 8;                // halves within 32-key row
    async_cp16(lv + ch * 16, vb + (long)row * SEQ + kt * 32 + off);
  }
}

// --------------------------- kernel 0: weight cvt ---------------------------
__global__ void k_prep(const float* __restrict__ qkv_w,
                       const float* __restrict__ proj_w,
                       _Float16* __restrict__ qw_h,
                       _Float16* __restrict__ pw_h) {
  const int i = blockIdx.x * blockDim.x + threadIdx.x;
  if (i < 3 * CH * CH) qw_h[i] = (_Float16)qkv_w[i];
  if (i < CH * CH)     pw_h[i] = (_Float16)proj_w[i];
}

// --------------------------- kernel 1: GroupNorm ----------------------------
__global__ void __launch_bounds__(256) k_groupnorm(
    const float* __restrict__ x, const float* __restrict__ gw,
    const float* __restrict__ gb, _Float16* __restrict__ z) {
  __shared__ float ssum[256];
  __shared__ float ssq[256];
  const int b  = blockIdx.x >> 5;
  const int g  = blockIdx.x & 31;
  const int c0 = g * CPG;
  const float* xg = x + ((long)(b * CH + c0) << 12);   // 4*4096 contiguous

  float s = 0.f, q = 0.f;
  for (int i = threadIdx.x; i < CPG * SEQ; i += 256) {
    const float v = xg[i];
    s += v;
    q += v * v;
  }
  ssum[threadIdx.x] = s;
  ssq[threadIdx.x]  = q;
  __syncthreads();
  for (int st = 128; st > 0; st >>= 1) {
    if (threadIdx.x < st) {
      ssum[threadIdx.x] += ssum[threadIdx.x + st];
      ssq[threadIdx.x]  += ssq[threadIdx.x + st];
    }
    __syncthreads();
  }
  const float mean = ssum[0] * (1.f / 16384.f);
  const float var  = ssq[0] * (1.f / 16384.f) - mean * mean;
  const float rstd = rsqrtf(var + GN_EPS);

  for (int i = threadIdx.x; i < CPG * SEQ; i += 256) {
    const int cc = i >> 12;
    const int n  = i & 4095;
    const int c  = c0 + cc;
    const float v = (xg[i] - mean) * rstd * gw[c] + gb[c];
    z[((long)((b << 12) + n)) * CH + c] = (_Float16)v;
  }
}

// --------------------------- kernel 2: QKV GEMM -----------------------------
__global__ void __launch_bounds__(256) k_qkv(
    const _Float16* __restrict__ z, const _Float16* __restrict__ wq,
    const float* __restrict__ bias, _Float16* __restrict__ q,
    _Float16* __restrict__ k, _Float16* __restrict__ vT) {
  const int lane = threadIdx.x & 31;
  const int id   = blockIdx.x * 8 + (threadIdx.x >> 5);
  const int nt   = id % 24;
  const int mt   = id / 24;
  const int row0 = mt * 16;
  const int col0 = nt * 16;

  v8f_t acc = {0.f, 0.f, 0.f, 0.f, 0.f, 0.f, 0.f, 0.f};
#pragma unroll
  for (int kc = 0; kc < 4; ++kc) {
    v16h_t a  = load_a16(z, CH, row0, kc * 32, lane);
    v16h_t bb = load_bt16(wq, CH, col0, kc * 32, lane);
    acc = wmma_f16(a, bb, acc);
  }

  const int seg = col0 >> 7;           // wave-uniform: 0=q 1=k 2=v
  const int col = col0 + (lane & 15);
  const float bv = bias[col];
  const int c   = col & 127;
  const int rhi = (lane & 16) ? 8 : 0;
#pragma unroll
  for (int r = 0; r < 8; ++r) {
    const int row = row0 + r + rhi;
    const float val = acc[r] + bv;
    if (seg == 0) {
      q[(long)row * CH + c] = (_Float16)val;
    } else if (seg == 1) {
      k[(long)row * CH + c] = (_Float16)val;
    } else {
      const int bidx = row >> 12;
      const int n    = row & 4095;
      vT[((long)(bidx * CH + c) << 12) + n] = (_Float16)val;
    }
  }
}

// --------------------------- kernel 3: flash attention ----------------------
// Block = 8 waves = 128 query rows; wave owns 16 rows. K/V 32-key tiles are
// staged block-wide into LDS with async global->LDS b128 DMA, double-buffered.
// Per tile/wave: 8 WMMA for S, shfl online softmax, P via wave-private LDS
// tile (D-layout -> A-layout), 8 WMMA for O += P*V.
__global__ void __launch_bounds__(256) k_attn(
    const _Float16* __restrict__ q, const _Float16* __restrict__ k,
    const _Float16* __restrict__ vT, _Float16* __restrict__ sa) {
  __shared__ __align__(16) _Float16 Ks[2][32 * CH];   // 2 x 8KB key tiles
  __shared__ __align__(16) _Float16 Vs[2][CH * 32];   // 2 x 8KB V^T tiles
  __shared__ __align__(16) _Float16 Pt[8][16 * 32];   // wave-private P tiles
  const int tid  = threadIdx.x;
  const int lane = tid & 31;
  const int w    = tid >> 5;
  const int b    = blockIdx.x >> 5;
  const int qt   = blockIdx.x & 31;
  const int row0 = (b << 12) + qt * 128 + w * 16;     // flat row in [16384]
  const _Float16* kb = k + ((long)(b << 12)) * CH;    // batch K rows, contig
  const _Float16* vb = vT + (((long)b * CH) << 12);   // batch V^T base
  _Float16* P = &Pt[w][0];

  v16h_t qa[4];
#pragma unroll
  for (int kc = 0; kc < 4; ++kc) qa[kc] = load_a16(q, CH, row0, kc * 32, lane);

  v8f_t o[8];
#pragma unroll
  for (int t = 0; t < 8; ++t) o[t] = (v8f_t){0.f,0.f,0.f,0.f,0.f,0.f,0.f,0.f};
  float mrow[8], lrow[8];
#pragma unroll
  for (int r = 0; r < 8; ++r) { mrow[r] = -3.0e38f; lrow[r] = 0.f; }

  const int rhi = (lane & 16) ? 8 : 0;
  const int cl  = lane & 15;

  stage_tile(kb, vb, &Ks[0][0], &Vs[0][0], 0, tid);
  wait_async0();
  __syncthreads();

  for (int kt = 0; kt < SEQ / 32; ++kt) {
    const int cb = kt & 1;
    if (kt + 1 < SEQ / 32) {           // overlap next tile's DMA with compute
      stage_tile(kb, vb, &Ks[cb ^ 1][0], &Vs[cb ^ 1][0], kt + 1, tid);
    }

    const _Float16* Kl = &Ks[cb][0];
    v8f_t sL = {0.f,0.f,0.f,0.f,0.f,0.f,0.f,0.f};
    v8f_t sR = {0.f,0.f,0.f,0.f,0.f,0.f,0.f,0.f};
#pragma unroll
    for (int kc = 0; kc < 4; ++kc) {
      v16h_t bL = load_bt16(Kl, CH, 0,  kc * 32, lane);
      v16h_t bR = load_bt16(Kl, CH, 16, kc * 32, lane);
      sL = wmma_f16(qa[kc], bL, sL);
      sR = wmma_f16(qa[kc], bR, sR);
    }

    float pl[8], pr[8];
#pragma unroll
    for (int r = 0; r < 8; ++r) {
      float sl = sL[r] * ATTN_SCALE;
      float sr = sR[r] * ATTN_SCALE;
      float mx = fmaxf(sl, sr);
#pragma unroll
      for (int d = 1; d < 16; d <<= 1) mx = fmaxf(mx, __shfl_xor(mx, d, 32));
      const float mnew = fmaxf(mrow[r], mx);
      const float el = __expf(sl - mnew);
      const float er = __expf(sr - mnew);
      float sum = el + er;
#pragma unroll
      for (int d = 1; d < 16; d <<= 1) sum += __shfl_xor(sum, d, 32);
      const float corr = __expf(mrow[r] - mnew);
      lrow[r] = lrow[r] * corr + sum;
      mrow[r] = mnew;
      pl[r] = el;
      pr[r] = er;
#pragma unroll
      for (int t = 0; t < 8; ++t) o[t][r] *= corr;
    }

    // D-layout -> LDS tile [16 rows][32 keys] (wave-private)
#pragma unroll
    for (int r = 0; r < 8; ++r) {
      P[(r + rhi) * 32 + cl]      = (_Float16)pl[r];
      P[(r + rhi) * 32 + 16 + cl] = (_Float16)pr[r];
    }
    __syncthreads();
    v16h_t pa = load_a16(P, 32, 0, 0, lane);      // A-fragment of P (16x32)
    const _Float16* Vl = &Vs[cb][0];
#pragma unroll
    for (int t = 0; t < 8; ++t) {
      v16h_t bv = load_bt16(Vl, 32, t * 16, 0, lane);
      o[t] = wmma_f16(pa, bv, o[t]);
    }
    wait_async0();        // next tile's DMA landed
    __syncthreads();      // everyone done reading cb before it is re-staged
  }

#pragma unroll
  for (int r = 0; r < 8; ++r) {
    const float inv = 1.f / lrow[r];
    const int row = row0 + r + rhi;
#pragma unroll
    for (int t = 0; t < 8; ++t) {
      sa[(long)row * CH + t * 16 + cl] = (_Float16)(o[t][r] * inv);
    }
  }
}

// --------------------------- kernel 4: proj + residual ----------------------
__global__ void __launch_bounds__(256) k_proj(
    const _Float16* __restrict__ sa, const _Float16* __restrict__ pw,
    const float* __restrict__ pb, const float* __restrict__ x,
    float* __restrict__ out) {
  const int lane = threadIdx.x & 31;
  const int id   = blockIdx.x * 8 + (threadIdx.x >> 5);
  const int nt   = id & 7;
  const int mt   = id >> 3;
  const int row0 = mt * 16;
  const int col0 = nt * 16;

  v8f_t acc = {0.f, 0.f, 0.f, 0.f, 0.f, 0.f, 0.f, 0.f};
#pragma unroll
  for (int kc = 0; kc < 4; ++kc) {
    v16h_t a  = load_a16(sa, CH, row0, kc * 32, lane);
    v16h_t bb = load_bt16(pw, CH, col0, kc * 32, lane);
    acc = wmma_f16(a, bb, acc);
  }

  const int c   = col0 + (lane & 15);
  const float bv = pb[c];
  const int rhi = (lane & 16) ? 8 : 0;
#pragma unroll
  for (int r = 0; r < 8; ++r) {
    const int row  = row0 + r + rhi;
    const int bidx = row >> 12;
    const int n    = row & 4095;
    const long idx = ((long)(bidx * CH + c) << 12) + n;   // [B][C][H*W]
    out[idx] = x[idx] + acc[r] + bv;
  }
}

// ---------------------------------------------------------------------------
extern "C" void kernel_launch(void* const* d_in, const int* in_sizes, int n_in,
                              void* d_out, int out_size, void* d_ws, size_t ws_size,
                              hipStream_t stream) {
  const float* x      = (const float*)d_in[0];
  const float* gn_w   = (const float*)d_in[1];
  const float* gn_b   = (const float*)d_in[2];
  const float* qkv_w  = (const float*)d_in[3];
  const float* qkv_b  = (const float*)d_in[4];
  const float* proj_w = (const float*)d_in[5];
  const float* proj_b = (const float*)d_in[6];
  float* out = (float*)d_out;

  _Float16* w   = (_Float16*)d_ws;
  _Float16* z   = w;                                  // [16384][128]
  _Float16* q   = z  + (size_t)ROWS * CH;             // [16384][128]
  _Float16* k   = q  + (size_t)ROWS * CH;             // [16384][128]
  _Float16* vT  = k  + (size_t)ROWS * CH;             // [4][128][4096]
  _Float16* sa  = vT + (size_t)ROWS * CH;             // [16384][128]
  _Float16* qwh = sa + (size_t)ROWS * CH;             // [384][128]
  _Float16* pwh = qwh + (size_t)3 * CH * CH;          // [128][128]

  k_prep<<<192, 256, 0, stream>>>(qkv_w, proj_w, qwh, pwh);
  k_groupnorm<<<BATCH * 32, 256, 0, stream>>>(x, gn_w, gn_b, z);
  k_qkv<<<(ROWS / 16) * 24 / 8, 256, 0, stream>>>(z, qwh, qkv_b, q, k, vT);
  k_attn<<<BATCH * (SEQ / 128), 256, 0, stream>>>(q, k, vT, sa);
  k_proj<<<(ROWS / 16) * 8 / 8, 256, 0, stream>>>(sa, pwh, proj_b, x, out);
}
